// AttentionDecoder_57097295233181
// MI455X (gfx1250) — matrix operand order repaired and checked by reading, not compiled, not run on hardware
//
#include <hip/hip_runtime.h>

// Problem constants
#define V_SZ 32000
#define E_SZ 256
#define H_SZ 512
#define L_SZ 2
#define B_SZ 128
#define S_SZ 512
#define XK   (H_SZ + E_SZ)   // 768 : width of x = [context | embedded]
#define G4H  (4 * H_SZ)      // 2048: gate width

typedef __attribute__((ext_vector_type(16))) __bf16        v16bf;
typedef __attribute__((ext_vector_type(8)))  unsigned int  v8u;
typedef __attribute__((ext_vector_type(8)))  float         v8f;

// ---------------------------------------------------------------- helpers

// Pack two fp32 into one dword of two bf16 (truncation) with a single
// v_perm_b32: take bytes [3:2] of lo into [1:0], bytes [3:2] of hi into [3:2].
__device__ __forceinline__ unsigned pk2(float lo, float hi) {
  return __builtin_amdgcn_perm(__builtin_bit_cast(unsigned, hi),
                               __builtin_bit_cast(unsigned, lo),
                               0x07060302u);
}

__device__ __forceinline__ v16bf pack16(float4 f0, float4 f1, float4 f2, float4 f3) {
  v8u u;
  u[0] = pk2(f0.x, f0.y); u[1] = pk2(f0.z, f0.w);
  u[2] = pk2(f1.x, f1.y); u[3] = pk2(f1.z, f1.w);
  u[4] = pk2(f2.x, f2.y); u[5] = pk2(f2.z, f2.w);
  u[6] = pk2(f3.x, f3.y); u[7] = pk2(f3.z, f3.w);
  return __builtin_bit_cast(v16bf, u);
}

// A fragment: 16x32 bf16 (MxK). lane l (g = l>>4, m = m0 + (l&15)):
//   elems 0..15 <- A[m][k0 + g*16 .. +15]   (16 consecutive K)
__device__ __forceinline__ v16bf load_a_frag(const float* __restrict__ rowk0, int g) {
  const float4* q0 = (const float4*)(rowk0 + g * 16);
  const float4* q1 = (const float4*)(rowk0 + g * 16 + 8);
  return pack16(q0[0], q0[1], q1[0], q1[1]);
}

// B fragment: 32x16 bf16 (KxN). lane l (g = l>>4, n = n0 + (l&15)):
//   elems 0..15 <- W[n][k0 + g*16 .. +15]   (16 consecutive K)
__device__ __forceinline__ v16bf load_b_frag(const float* __restrict__ rowk0g) {
  const float4* q = (const float4*)rowk0g;
  return pack16(q[0], q[1], q[2], q[3]);
}

__device__ __forceinline__ float wave_sum(float v) {
  #pragma unroll
  for (int off = 16; off > 0; off >>= 1) v += __shfl_xor(v, off, 32);
  return v;
}

__device__ __forceinline__ float sigmoidf(float x) {
  return 1.0f / (1.0f + __expf(-x));
}

// ---------------------------------------------------------------- kernels

// x[b][H + e] = emb[dec_input[b]][e]
__global__ void embed_kernel(const int* __restrict__ dec,
                             const float* __restrict__ emb,
                             float* __restrict__ x) {
  int b = blockIdx.x, e = threadIdx.x;                // 256 threads == E
  x[(size_t)b * XK + H_SZ + e] = emb[(size_t)dec[b] * E_SZ + e];
}

// hdot[b] = dot(hidden0[b], attn_w[H:2H])
__global__ void hdot_kernel(const float* __restrict__ hidden,
                            const float* __restrict__ attn_w,
                            float* __restrict__ hdot) {
  int wid = threadIdx.x >> 5, lane = threadIdx.x & 31;
  int b = blockIdx.x * 8 + wid;                       // 16 blocks x 8 waves = 128
  float acc = 0.f;
  for (int h = lane; h < H_SZ; h += 32)
    acc += hidden[(size_t)b * H_SZ + h] * attn_w[H_SZ + h];
  acc = wave_sum(acc);
  if (lane == 0) hdot[b] = acc;
}

// logits[b][s] = dot(enc[s][b][:], attn_w[0:H]) + hdot[b] + attn_b
__global__ void logits_kernel(const float* __restrict__ enc,
                              const float* __restrict__ attn_w,
                              const float* __restrict__ attn_b,
                              const float* __restrict__ hdot,
                              float* __restrict__ logits) {
  int wid = threadIdx.x >> 5, lane = threadIdx.x & 31;
  int p = blockIdx.x * 8 + wid;                       // p in [0, B*S)
  int b = p >> 9, s = p & (S_SZ - 1);
  const float* row = enc + ((size_t)s * B_SZ + b) * H_SZ;
  float acc = 0.f;
  for (int h = lane; h < H_SZ; h += 32) acc += row[h] * attn_w[h];
  acc = wave_sum(acc);
  if (lane == 0) logits[(size_t)b * S_SZ + s] = acc + hdot[b] + attn_b[0];
}

// in-place softmax over S per batch row (block per b, 256 threads, 2 elems each)
__global__ void softmax_kernel(float* __restrict__ logits) {
  __shared__ float red[256];
  int b = blockIdx.x, t = threadIdx.x;
  float* row = logits + (size_t)b * S_SZ;
  float v0 = row[t], v1 = row[t + 256];
  red[t] = fmaxf(v0, v1); __syncthreads();
  for (int s = 128; s > 0; s >>= 1) { if (t < s) red[t] = fmaxf(red[t], red[t + s]); __syncthreads(); }
  float m = red[0]; __syncthreads();
  float e0 = __expf(v0 - m), e1 = __expf(v1 - m);
  red[t] = e0 + e1; __syncthreads();
  for (int s = 128; s > 0; s >>= 1) { if (t < s) red[t] += red[t + s]; __syncthreads(); }
  float inv = 1.0f / red[0];
  row[t] = e0 * inv; row[t + 256] = e1 * inv;
}

// x[b][h] = sum_s alpha[b][s] * enc[s][b][h]   (context)
__global__ void context_kernel(const float* __restrict__ alpha,
                               const float* __restrict__ enc,
                               float* __restrict__ x) {
  __shared__ float sal[S_SZ];
  int b = blockIdx.y, t = threadIdx.x;
  int h = blockIdx.x * 256 + t;
  sal[t] = alpha[(size_t)b * S_SZ + t];
  sal[t + 256] = alpha[(size_t)b * S_SZ + t + 256];
  __syncthreads();
  float acc = 0.f;
  for (int s = 0; s < S_SZ; ++s)
    acc += sal[s] * enc[((size_t)s * B_SZ + b) * H_SZ + h];
  x[(size_t)b * XK + h] = acc;
}

// C[128 x N] (+)= A[128 x K] @ W[N x K]^T  + b1 + b2, bf16 WMMA, fp32 accum.
// grid.x = N/32, block = (32,2). Each wave owns a 64(M) x 32(N) macro-tile:
// 4 M-tiles x 2 N-tiles; each A fragment is reused for two WMMAs, and the two
// independent WMMAs per A-frag fill the WMMA->VALU hazard slots.
template <int K>
__global__ void __launch_bounds__(64)
gemm128_wmma_bf16(const float* __restrict__ A, const float* __restrict__ W,
                  const float* __restrict__ b1, const float* __restrict__ b2,
                  float* __restrict__ C, int N, int accum) {
  const int n0   = blockIdx.x << 5;      // 32 columns per block
  const int lane = threadIdx.x;
  const int wid  = threadIdx.y;          // 0..1
  const int nl   = lane & 15;
  const int g    = lane >> 4;            // K-half selector

  v8f acc[4][2];
  #pragma unroll
  for (int i = 0; i < 4; ++i)
    #pragma unroll
    for (int j = 0; j < 2; ++j) acc[i][j] = (v8f)(0.0f);

  const float* wrow0 = W + (size_t)(n0 + nl) * K + g * 16;        // n-tile 0
  const float* wrow1 = wrow0 + (size_t)16 * K;                    // n-tile 1
  const float* arow  = A + (size_t)nl * K;

  for (int k0 = 0; k0 < K; k0 += 32) {
    v16bf bfrag0 = load_b_frag(wrow0 + k0);
    v16bf bfrag1 = load_b_frag(wrow1 + k0);
    #pragma unroll
    for (int mt = 0; mt < 4; ++mt) {
      const int mtile = (wid << 2) + mt;
      v16bf afrag = load_a_frag(arow + (size_t)(mtile << 4) * K + k0, g);
      acc[mt][0] = __builtin_amdgcn_wmma_f32_16x16x32_bf16(
          false, afrag, false, bfrag0, (short)0, acc[mt][0], false, false);
      acc[mt][1] = __builtin_amdgcn_wmma_f32_16x16x32_bf16(
          false, afrag, false, bfrag1, (short)0, acc[mt][1], false, false);
    }
  }

  float bias[2] = {0.f, 0.f};
  #pragma unroll
  for (int nt = 0; nt < 2; ++nt) {
    if (b1) bias[nt] += b1[n0 + (nt << 4) + nl];
    if (b2) bias[nt] += b2[n0 + (nt << 4) + nl];
  }

  const int mhi = g * 8;
  #pragma unroll
  for (int mt = 0; mt < 4; ++mt) {
    const int mbase = ((wid << 2) + mt) << 4;
    #pragma unroll
    for (int nt = 0; nt < 2; ++nt) {
      #pragma unroll
      for (int r = 0; r < 8; ++r) {
        size_t idx = (size_t)(mbase + mhi + r) * N + n0 + (nt << 4) + nl;
        float v = acc[mt][nt][r] + bias[nt];
        if (accum) v += C[idx];
        C[idx] = v;
      }
    }
  }
}

// LSTM pointwise: gates -> (h2, c2)
__global__ void lstm_pointwise(const float* __restrict__ gates,
                               const float* __restrict__ c_in,
                               float* __restrict__ h_out,
                               float* __restrict__ c_out) {
  int idx = blockIdx.x * 256 + threadIdx.x;           // [0, B*H)
  int b = idx >> 9, j = idx & (H_SZ - 1);
  const float* gr = gates + (size_t)b * G4H;
  float i = sigmoidf(gr[j]);
  float f = sigmoidf(gr[H_SZ + j]);
  float gg = tanhf(gr[2 * H_SZ + j]);
  float o = sigmoidf(gr[3 * H_SZ + j]);
  float c2 = f * c_in[idx] + i * gg;
  h_out[idx] = o * tanhf(c2);
  c_out[idx] = c2;
}

// ---------------------------------------------------------------- launch
extern "C" void kernel_launch(void* const* d_in, const int* in_sizes, int n_in,
                              void* d_out, int out_size, void* d_ws, size_t ws_size,
                              hipStream_t stream) {
  const int*   dec     = (const int*)  d_in[0];
  const float* hidden  = (const float*)d_in[1];   // (L,B,H)
  const float* cell    = (const float*)d_in[2];   // (L,B,H)
  const float* enc     = (const float*)d_in[3];   // (S,B,H)
  const float* emb     = (const float*)d_in[4];   // (V,E)
  const float* attn_w  = (const float*)d_in[5];   // (1,2H)
  const float* attn_b  = (const float*)d_in[6];   // (1,)
  const float* w_ih0   = (const float*)d_in[7];   // (4H, E+H)
  const float* w_hh0   = (const float*)d_in[8];   // (4H, H)
  const float* b_ih0   = (const float*)d_in[9];
  const float* b_hh0   = (const float*)d_in[10];
  const float* w_ih1   = (const float*)d_in[11];  // (4H, H)
  const float* w_hh1   = (const float*)d_in[12];  // (4H, H)
  const float* b_ih1   = (const float*)d_in[13];
  const float* b_hh1   = (const float*)d_in[14];
  const float* fc_w    = (const float*)d_in[15];  // (V, H)
  const float* fc_b    = (const float*)d_in[16];  // (V,)

  // outputs: [prediction (B*V) | new_hidden (L*B*H) | new_cell (L*B*H)]
  float* out_pred = (float*)d_out;
  float* out_hid  = out_pred + (size_t)B_SZ * V_SZ;
  float* out_cel  = out_hid + (size_t)L_SZ * B_SZ * H_SZ;
  float* h0_new = out_hid;                 // layer 0
  float* h1_new = out_hid + B_SZ * H_SZ;   // layer 1
  float* c0_new = out_cel;
  float* c1_new = out_cel + B_SZ * H_SZ;

  // workspace layout (floats)
  float* ws      = (float*)d_ws;
  float* x       = ws;                               // B * XK      = 98304
  float* logits  = x + (size_t)B_SZ * XK;            // B * S       = 65536
  float* hdot    = logits + (size_t)B_SZ * S_SZ;     // B           = 128
  float* gates0  = hdot + B_SZ;                      // B * 4H      = 262144
  float* gates1  = gates0 + (size_t)B_SZ * G4H;      // B * 4H      = 262144

  // 1) embedding gather into x[:, H:H+E]
  embed_kernel<<<B_SZ, E_SZ, 0, stream>>>(dec, emb, x);
  // 2) hdot[b] = h0[b] . w_hid
  hdot_kernel<<<B_SZ / 8, 256, 0, stream>>>(hidden, attn_w, hdot);
  // 3) attention logits
  logits_kernel<<<(B_SZ * S_SZ) / 8, 256, 0, stream>>>(enc, attn_w, attn_b, hdot, logits);
  // 4) softmax over S (in place -> alpha)
  softmax_kernel<<<B_SZ, 256, 0, stream>>>(logits);
  // 5) context into x[:, 0:H]
  context_kernel<<<dim3(H_SZ / 256, B_SZ), 256, 0, stream>>>(logits, enc, x);

  dim3 gblk(32, 2);
  // 6) gates0 = x @ w_ih0^T + b_ih0 + b_hh0 ; then += h0 @ w_hh0^T
  gemm128_wmma_bf16<XK><<<G4H / 32, gblk, 0, stream>>>(x, w_ih0, b_ih0, b_hh0, gates0, G4H, 0);
  gemm128_wmma_bf16<H_SZ><<<G4H / 32, gblk, 0, stream>>>(hidden, w_hh0, nullptr, nullptr, gates0, G4H, 1);
  // 7) LSTM layer 0 pointwise -> h0_new, c0_new
  lstm_pointwise<<<(B_SZ * H_SZ) / 256, 256, 0, stream>>>(gates0, cell, h0_new, c0_new);
  // 8) gates1 = h0_new @ w_ih1^T + b_ih1 + b_hh1 ; then += h1_old @ w_hh1^T
  gemm128_wmma_bf16<H_SZ><<<G4H / 32, gblk, 0, stream>>>(h0_new, w_ih1, b_ih1, b_hh1, gates1, G4H, 0);
  gemm128_wmma_bf16<H_SZ><<<G4H / 32, gblk, 0, stream>>>(hidden + (size_t)B_SZ * H_SZ, w_hh1,
                                                         nullptr, nullptr, gates1, G4H, 1);
  // 9) LSTM layer 1 pointwise -> h1_new, c1_new
  lstm_pointwise<<<(B_SZ * H_SZ) / 256, 256, 0, stream>>>(gates1, cell + (size_t)B_SZ * H_SZ,
                                                          h1_new, c1_new);
  // 10) prediction = h1_new @ fc_w^T + fc_b   (the big, HBM-bound GEMM)
  gemm128_wmma_bf16<H_SZ><<<V_SZ / 32, gblk, 0, stream>>>(h1_new, fc_w, fc_b, nullptr, out_pred,
                                                          V_SZ, 0);
}
